// ConvLayer_86294482912121
// MI455X (gfx1250) — matrix operand order repaired and checked by reading, not compiled, not run on hardware
//
#include <hip/hip_runtime.h>

// ---------------- problem constants ----------------
#define BB   16
#define NN   256
#define MM   12
#define HAh  64
#define HBh  64
#define HH   4
#define DIN  192          // 2*HA + HB
#define DOUT 1024         // 4*H*HA
#define HHA  256          // H*HA

typedef __attribute__((ext_vector_type(16))) __bf16 v16bf;
typedef __attribute__((ext_vector_type(8)))  __bf16 v8bf;
typedef __attribute__((ext_vector_type(8)))  float  v8f;

// ---------------- workspace layout (bytes) ----------------
#define WS_S1    0                         // f32 [B*N*M*64]  = 12582912 B
#define WS_S3    12582912                  // f32 [B*N*64]    =  1048576 B
#define WS_STAT  13631488                  // f32 [256]
#define WS_WB    13632512                  // bf16 frags big GEMM: 64ct*6kt*32lane*16 = 393216 B
#define WS_WF    14025728                  // bf16 frags final GEMM: 4ct*8kt*32lane*16 = 32768 B

// A-matrix fragment load, contiguous K order (matches packed B):
// lanes 0-15: elems 0..15 -> K=0..15
// lanes 16-31: elems 0..15 -> K=16..31
__device__ inline v16bf load_a_frag(const __bf16* p, int lane) {
    int ko = (lane >> 4) * 16;
    v8bf lo = *(const v8bf*)(p + ko);
    v8bf hi = *(const v8bf*)(p + ko + 8);
    v16bf a;
#pragma unroll
    for (int e = 0; e < 8; ++e) { a[e] = lo[e]; a[8 + e] = hi[e]; }
    return a;
}

__device__ inline v8f wmma_bf16(v16bf a, v16bf b, v8f c) {
    return __builtin_amdgcn_wmma_f32_16x16x32_bf16(false, a, false, b, (short)0, c, false, false);
}

// ---------------- kernel 1: pack weights into B-fragment layout ----------------
// B-fragment layout packed here (contiguous K, same as load_a_frag): lane = (khalf<<4)|col ; element e -> K = khalf*16+e, N = col.
__global__ void pack_weights(const float* __restrict__ Wfull,
                             const float* __restrict__ Wfin,
                             __bf16* __restrict__ Wb, __bf16* __restrict__ Wf) {
    int t = blockIdx.x * blockDim.x + threadIdx.x;
    const int NB = 64 * 6 * 32;
    const int NF = 4 * 8 * 32;
    if (t < NB) {
        int lane = t & 31;
        int kt   = (t >> 5) % 6;
        int ct   = t / 192;
        int col  = ct * 16 + (lane & 15);
        int kb   = kt * 32 + ((lane >> 4) << 4);
        __bf16* dst = Wb + t * 16;
#pragma unroll
        for (int e = 0; e < 16; ++e) dst[e] = (__bf16)Wfull[(kb + e) * DOUT + col];
    } else if (t < NB + NF) {
        int u = t - NB;
        int lane = u & 31;
        int kt   = (u >> 5) & 7;
        int ct   = u >> 8;
        int col  = ct * 16 + (lane & 15);
        int kb   = kt * 32 + ((lane >> 4) << 4);
        __bf16* dst = Wf + u * 16;
#pragma unroll
        for (int e = 0; e < 16; ++e) dst[e] = (__bf16)Wfin[(kb + e) * HAh + col];
    }
}

__global__ void zero_stats(float* stats) { stats[threadIdx.x] = 0.0f; }

// ---------------- kernel 2: fused per-atom pipeline, 2 atoms per workgroup ----------------
__global__ __launch_bounds__(256)
void conv_main(const float* __restrict__ atom_emb,
               const float* __restrict__ nbr_emb,
               const float* __restrict__ atom_mask,
               const int*   __restrict__ adj,
               const float* __restrict__ b_full,
               const float* __restrict__ b_final,
               const __bf16* __restrict__ Wb,
               const __bf16* __restrict__ Wf,
               float* __restrict__ S1) {
    const int bn0  = blockIdx.x * 2;       // two consecutive atoms (same batch b)
    const int bId  = bn0 >> 8;             // N = 256
    const int tid  = threadIdx.x;
    const int lane = tid & 31;
    const int wave = tid >> 5;

    // X tile is dead after phase 1; Y (gated values) reuses its LDS.
    __shared__ __align__(16) union {
        __bf16 X[2][16][208];              // concat input, K padded       (13312 B)
        __bf16 Y[2][16][HHA];              // gated values for final GEMM  (16384 B)
    } XY;
    __shared__ __align__(16) __bf16 TGs[2][4][16][HHA];  // gate/core/a1/a2 (65536 B)
    __shared__ float Ssc[2][HH][MM][MM];                 // scores/probs     (4608 B)

    float mk[2] = { atom_mask[bn0], atom_mask[bn0 + 1] };

    // ---- phase 0: build X = [self | nbr_atom*mask | bond] rows 0..11, zero pad ----
    for (int idx = tid; idx < 2 * 16 * DIN; idx += 256) {
        int a = idx / (16 * DIN);
        int rem = idx % (16 * DIN);
        int m = rem / DIN, c = rem % DIN;
        int bn = bn0 + a;
        float v = 0.0f;
        if (m < MM) {
            if (c < HAh) {
                v = atom_emb[bn * HAh + c];
            } else if (c < 2 * HAh) {
                int nb = adj[bn * MM + m];
                v = atom_emb[(bId * NN + nb) * HAh + (c - HAh)] * mk[a];
            } else {
                v = nbr_emb[(bn * MM + m) * HBh + (c - 2 * HAh)];
            }
        }
        XY.X[a][m][c] = (__bf16)v;
    }
    __syncthreads();

    // ---- phase 1: TG = X[16x192] @ W_full[192x1024] + b_full  (WMMA bf16) ----
    // A fragments for both atoms hoisted to registers; reused across all 8 col tiles.
    v16bf afr[2][6];
#pragma unroll
    for (int a = 0; a < 2; ++a)
#pragma unroll
        for (int kt = 0; kt < 6; ++kt)
            afr[a][kt] = load_a_frag(&XY.X[a][lane & 15][kt * 32], lane);

#pragma unroll
    for (int i = 0; i < 8; ++i) {
        int ct = wave * 8 + i;                 // 64 column tiles of 16
        if (i < 7)  // prefetch next tile's B fragments (-> global_prefetch_b8)
            __builtin_prefetch(Wb + (((ct + 1) * 6) * 32 + lane) * 16, 0, 1);
        float binit = b_full[ct * 16 + (lane & 15)];
        v8f acc0, acc1;
#pragma unroll
        for (int r = 0; r < 8; ++r) { acc0[r] = binit; acc1[r] = binit; }
#pragma unroll
        for (int kt = 0; kt < 6; ++kt) {
            v16bf b = *(const v16bf*)(Wb + ((ct * 6 + kt) * 32 + lane) * 16);
            acc0 = wmma_bf16(afr[0][kt], b, acc0);
            acc1 = wmma_bf16(afr[1][kt], b, acc1);
        }
        int q = ct >> 4, col = (ct & 15) * 16 + (lane & 15);
        int mbase = (lane < 16) ? 0 : 8;
#pragma unroll
        for (int r = 0; r < 8; ++r) {
            TGs[0][q][mbase + r][col] = (__bf16)acc0[r];
            TGs[1][q][mbase + r][col] = (__bf16)acc1[r];
        }
    }
    __syncthreads();

    // ---- phase 2: attention scores s[h][m][k] = a1[m,h,:]·a2[k,h,:]/sqrt(H) + bias ----
    for (int idx = tid; idx < 2 * HH * MM * MM; idx += 256) {
        int a = idx / (HH * MM * MM);
        int rem = idx % (HH * MM * MM);
        int h = rem / (MM * MM);
        int rm = rem % (MM * MM);
        int m = rm / MM, k = rm % MM;
        const __bf16* pa = &TGs[a][2][m][h * HAh];
        const __bf16* pb = &TGs[a][3][k][h * HAh];
        float dot = 0.0f;
#pragma unroll
        for (int d = 0; d < HAh; ++d) dot += (float)pa[d] * (float)pb[d];
        Ssc[a][h][m][k] = dot * 0.5f + (1.0f - mk[a]) * -10000.0f;  // 1/sqrt(4)=0.5
    }
    __syncthreads();

    // ---- phase 3: softmax over k ----
    if (tid < 2 * HH * MM) {
        int a = tid / (HH * MM);
        int rem = tid % (HH * MM);
        int h = rem / MM, m = rem % MM;
        float mx = -1e30f;
        for (int k = 0; k < MM; ++k) mx = fmaxf(mx, Ssc[a][h][m][k]);
        float s = 0.0f;
        for (int k = 0; k < MM; ++k) { float e = __expf(Ssc[a][h][m][k] - mx); Ssc[a][h][m][k] = e; s += e; }
        float inv = 1.0f / s;
        for (int k = 0; k < MM; ++k) Ssc[a][h][m][k] *= inv;
    }
    __syncthreads();

    // ---- phase 4: ctx = probs @ core ; Y = sigmoid(gate)*relu(ctx) ----
    for (int idx = tid; idx < 2 * MM * HHA; idx += 256) {
        int a = idx / (MM * HHA);
        int rem = idx % (MM * HHA);
        int m = rem >> 8, c = rem & 255, h = c >> 6;
        float ctx = 0.0f;
#pragma unroll
        for (int k = 0; k < MM; ++k) ctx += Ssc[a][h][m][k] * (float)TGs[a][1][k][c];
        float g = (float)TGs[a][0][m][c];
        float sg = 1.0f / (1.0f + __expf(-g));
        XY.Y[a][m][c] = (__bf16)(sg * fmaxf(ctx, 0.0f));
    }
#pragma unroll
    for (int a = 0; a < 2; ++a)
#pragma unroll
        for (int j = 0; j < 4; ++j) XY.Y[a][MM + j][tid] = (__bf16)0.0f;   // pad rows
    __syncthreads();

    // ---- phase 5: S1 = Y[16x256] @ W_final[256x64] + b_final (WMMA bf16) ----
    // 8 waves = 2 atoms x 4 column tiles
    {
        int a  = wave >> 2;
        int ct = wave & 3;
        int bn = bn0 + a;
        float binit = b_final[ct * 16 + (lane & 15)];
        v8f acc;
#pragma unroll
        for (int r = 0; r < 8; ++r) acc[r] = binit;
#pragma unroll
        for (int kt = 0; kt < 8; ++kt) {
            v16bf af = load_a_frag(&XY.Y[a][lane & 15][kt * 32], lane);
            v16bf b  = *(const v16bf*)(Wf + ((ct * 8 + kt) * 32 + lane) * 16);
            acc = wmma_bf16(af, b, acc);
        }
        int col = ct * 16 + (lane & 15);
        int mbase = (lane < 16) ? 0 : 8;
#pragma unroll
        for (int r = 0; r < 8; ++r) {
            int m = mbase + r;
            if (m < MM) S1[((long)bn * MM + m) * HAh + col] = acc[r];
        }
    }
}

// ---------------- kernel 3: channel sum / sumsq reduction (64 channels) ----------------
__global__ void reduce_stats(const float* __restrict__ src, int rows, float* __restrict__ sums) {
    int c = threadIdx.x & 63;
    int walker = blockIdx.x * (blockDim.x >> 6) + (threadIdx.x >> 6);
    int nwalk  = gridDim.x * (blockDim.x >> 6);
    float s = 0.0f, q = 0.0f;
    for (int r = walker; r < rows; r += nwalk) {
        float v = src[(long)r * 64 + c];
        s += v; q += v * v;
    }
    atomicAdd(&sums[c], s);
    atomicAdd(&sums[64 + c], q);
}

// ---------------- kernel 4: BN1 + relu + sum over neighbors ----------------
__global__ void bn1_sum(const float* __restrict__ S1, const float* __restrict__ st,
                        const float* __restrict__ g, const float* __restrict__ b,
                        float* __restrict__ S3) {
    int idx = blockIdx.x * blockDim.x + threadIdx.x;
    if (idx >= BB * NN * HAh) return;
    int c = idx & 63, bn = idx >> 6;
    const float R = (float)(BB * NN * MM);
    float mean = st[c] / R;
    float var  = st[64 + c] / R - mean * mean;
    float scale = g[c] * rsqrtf(var + 1e-5f);
    float bias  = b[c];
    float acc = 0.0f;
#pragma unroll
    for (int m = 0; m < MM; ++m) {
        float v = S1[((long)bn * MM + m) * 64 + c];
        acc += fmaxf((v - mean) * scale + bias, 0.0f);
    }
    S3[idx] = acc;
}

// ---------------- kernel 5: BN2 + residual + relu ----------------
__global__ void bn2_final(const float* __restrict__ S3, const float* __restrict__ st,
                          const float* __restrict__ g, const float* __restrict__ b,
                          const float* __restrict__ atom_emb, float* __restrict__ out) {
    int idx = blockIdx.x * blockDim.x + threadIdx.x;
    if (idx >= BB * NN * HAh) return;
    int c = idx & 63;
    const float R = (float)(BB * NN);
    float mean = st[c] / R;
    float var  = st[64 + c] / R - mean * mean;
    float scale = g[c] * rsqrtf(var + 1e-5f);
    float v = (S3[idx] - mean) * scale + b[c];
    out[idx] = fmaxf(atom_emb[idx] + v, 0.0f);
}

// ---------------- host launcher ----------------
extern "C" void kernel_launch(void* const* d_in, const int* in_sizes, int n_in,
                              void* d_out, int out_size, void* d_ws, size_t ws_size,
                              hipStream_t stream) {
    const float* atom_emb  = (const float*)d_in[0];
    const float* nbr_emb   = (const float*)d_in[1];
    const float* atom_mask = (const float*)d_in[2];
    const float* W_full    = (const float*)d_in[3];
    const float* b_full    = (const float*)d_in[4];
    const float* W_final   = (const float*)d_in[5];
    const float* b_final   = (const float*)d_in[6];
    const float* bn1_g     = (const float*)d_in[7];
    const float* bn1_b     = (const float*)d_in[8];
    const float* bn2_g     = (const float*)d_in[9];
    const float* bn2_b     = (const float*)d_in[10];
    const int*   adj       = (const int*)d_in[11];

    char* ws = (char*)d_ws;
    float*  S1    = (float*)(ws + WS_S1);
    float*  S3    = (float*)(ws + WS_S3);
    float*  stats = (float*)(ws + WS_STAT);
    __bf16* Wb    = (__bf16*)(ws + WS_WB);
    __bf16* Wf    = (__bf16*)(ws + WS_WF);

    pack_weights<<<52, 256, 0, stream>>>(W_full, W_final, Wb, Wf);
    zero_stats<<<1, 256, 0, stream>>>(stats);
    conv_main<<<(BB * NN) / 2, 256, 0, stream>>>(atom_emb, nbr_emb, atom_mask, adj,
                                                 b_full, b_final, Wb, Wf, S1);
    reduce_stats<<<192, 256, 0, stream>>>(S1, BB * NN * MM, stats);
    bn1_sum<<<(BB * NN * HAh + 255) / 256, 256, 0, stream>>>(S1, stats, bn1_g, bn1_b, S3);
    reduce_stats<<<64, 256, 0, stream>>>(S3, BB * NN, stats + 128);
    bn2_final<<<(BB * NN * HAh + 255) / 256, 256, 0, stream>>>(S3, stats + 128, bn2_g, bn2_b,
                                                               atom_emb, (float*)d_out);
}
